// WeightingAttn_3624952398660
// MI455X (gfx1250) — matrix operand order repaired and checked by reading, not compiled, not run on hardware
//
#include <hip/hip_runtime.h>

typedef __bf16 bf16_t;
typedef __attribute__((ext_vector_type(16))) __bf16 v16bf;
typedef __attribute__((ext_vector_type(8)))  __bf16 v8bf;
typedef __attribute__((ext_vector_type(8)))  float  v8f;

#define BATCH 8
#define SEQ   2048
#define DMODEL 1024
#define EDIM  16

static __device__ __forceinline__ v16bf zero_v16bf() {
    v16bf z;
    #pragma unroll
    for (int h = 0; h < 16; ++h) z[h] = (bf16_t)0.0f;
    return z;
}

// ---------------------------------------------------------------------------
// Kernel 0: one-shot fp32 -> bf16 conversion of w_q / w_k (16x1024 each).
// ---------------------------------------------------------------------------
__global__ __launch_bounds__(256) void cvt_w(
    const float* __restrict__ wq, const float* __restrict__ wk,
    bf16_t* __restrict__ Wq, bf16_t* __restrict__ Wk)
{
    const int i = blockIdx.x * 256 + threadIdx.x;   // 0 .. 16383
    Wq[i] = (bf16_t)wq[i];
    Wk[i] = (bf16_t)wk[i];
}

// ---------------------------------------------------------------------------
// Kernel 1: Q/K projection with WMMA.  One wave per 16-row tile.
// Q[b,s,16] = x @ w_q^T + b_q   (stored bf16), same for K.
// ---------------------------------------------------------------------------
__global__ __launch_bounds__(256) void proj_qk(
    const float* __restrict__ x,
    const bf16_t* __restrict__ Wq, const float* __restrict__ bq,
    const bf16_t* __restrict__ Wk, const float* __restrict__ bk,
    bf16_t* __restrict__ Qb, bf16_t* __restrict__ Kb)
{
    const int lane = threadIdx.x & 31;
    const int wave = threadIdx.x >> 5;
    const int l16  = lane & 15;
    const int half = lane >> 4;
    const long tile = (long)blockIdx.x * 8 + wave;   // 0..1023
    const long row0 = tile * 16;                     // flat row in [0, B*S)

    const float* xrow = x + (row0 + l16) * DMODEL;   // A row = l16
    // B operand base: lanes 0..15 take K=k0+h, lanes 16..31 take K=k0+16+h
    const bf16_t* wqrow = Wq + (long)l16 * DMODEL + half * 16;
    const bf16_t* wkrow = Wk + (long)l16 * DMODEL + half * 16;

    v8f cq; v8f ck;
    #pragma unroll
    for (int r = 0; r < 8; ++r) { cq[r] = 0.0f; ck[r] = 0.0f; }

    for (int k0 = 0; k0 < DMODEL; k0 += 32) {
        // A operand: x tile 16x32 bf16.  halves 0..7 -> K = k0+half*16+h,
        // halves 8..15 -> K = k0+half*16+8+(h-8)
        v16bf a;
        const float* p0 = xrow + k0 + half * 16;
        const float* p1 = xrow + k0 + half * 16 + 8;
        #pragma unroll
        for (int h = 0; h < 8; ++h) {
            a[h]     = (bf16_t)p0[h];
            a[8 + h] = (bf16_t)p1[h];
        }
        const v16bf bwq = *(const v16bf*)(wqrow + k0);
        const v16bf bwk = *(const v16bf*)(wkrow + k0);
        cq = __builtin_amdgcn_wmma_f32_16x16x32_bf16(false, a, false, bwq,
                                                     (short)0, cq, false, false);
        ck = __builtin_amdgcn_wmma_f32_16x16x32_bf16(false, a, false, bwk,
                                                     (short)0, ck, false, false);
    }

    const float biasq = bq[l16];
    const float biask = bk[l16];
    #pragma unroll
    for (int r = 0; r < 8; ++r) {
        const int row = r + half * 8;              // C layout: M = r + 8*(lane>=16)
        Qb[(row0 + row) * EDIM + l16] = (bf16_t)(cq[r] + biasq);
        Kb[(row0 + row) * EDIM + l16] = (bf16_t)(ck[r] + biask);
    }
}

// ---------------------------------------------------------------------------
// Kernel 2: x fp32 [b,s,d]  ->  xT bf16 [b,d,s]   (64x64 LDS tiles)
// Gives the flash kernel contiguous key-major 32B loads for the PV B-operand.
// ---------------------------------------------------------------------------
__global__ __launch_bounds__(256) void transpose_x(
    const float* __restrict__ x, bf16_t* __restrict__ XT)
{
    __shared__ bf16_t tile[64][65];
    const int b  = blockIdx.z;
    const int s0 = blockIdx.x * 64;
    const int d0 = blockIdx.y * 64;
    const int t  = threadIdx.x;

    const int rr = t >> 4;           // 0..15
    const int c4 = (t & 15) * 4;     // 0..60
    const float* xp = x + ((long)b * SEQ + s0) * DMODEL + d0;
    #pragma unroll
    for (int j = 0; j < 4; ++j) {
        const int row = rr + j * 16;
        const float* p = xp + (long)row * DMODEL + c4;
        #pragma unroll
        for (int i = 0; i < 4; ++i) tile[row][c4 + i] = (bf16_t)p[i];
    }
    __syncthreads();
    const int col = t >> 2;          // 0..63
    const int sr0 = (t & 3) * 16;    // 0,16,32,48
    bf16_t* op = XT + ((long)b * DMODEL + d0 + col) * SEQ + s0 + sr0;
    #pragma unroll
    for (int i = 0; i < 16; ++i) op[i] = tile[sr0 + i][col];
}

// ---------------------------------------------------------------------------
// 32-key block body for the attention kernel.  MASKED only for the single
// diagonal block (the 16-row diagonal always sits inside one 32-key block).
// ---------------------------------------------------------------------------
template <bool MASKED>
static __device__ __forceinline__ void kblock(
    const int k0, const int qt, const int l16, const int half,
    bf16_t (* __restrict__ pb)[32],
    const bf16_t* __restrict__ kbase, const bf16_t* __restrict__ xbase,
    const v16bf& aq, float* __restrict__ lsum, v8f* __restrict__ acc)
{
    // ---- scores: two 16x16 subtiles -----------------------------------
    v16bf bk0 = zero_v16bf();
    v16bf bk1 = zero_v16bf();
    if (half == 0) {                   // lanes 0..15 carry e=0..15; rest pad
        bk0 = *(const v16bf*)(kbase + (long)k0 * EDIM);
        bk1 = *(const v16bf*)(kbase + (long)(k0 + 16) * EDIM);
    }
    v8f s0, s1;
    #pragma unroll
    for (int r = 0; r < 8; ++r) { s0[r] = 0.0f; s1[r] = 0.0f; }
    s0 = __builtin_amdgcn_wmma_f32_16x16x32_bf16(false, aq, false, bk0,
                                                 (short)0, s0, false, false);
    s1 = __builtin_amdgcn_wmma_f32_16x16x32_bf16(false, aq, false, bk1,
                                                 (short)0, s1, false, false);

    // scale 1/sqrt(16) folded with log2(e): exp() becomes raw v_exp_f32
    const float SC = 0.25f * 1.44269504088896340736f;

    // ---- exp (+ causal mask on diagonal block); per-lane partial sums -
    #pragma unroll
    for (int r = 0; r < 8; ++r) {
        float p0 = __builtin_amdgcn_exp2f(s0[r] * SC);
        float p1 = __builtin_amdgcn_exp2f(s1[r] * SC);
        if (MASKED) {
            const int rowg = qt * 16 + r + half * 8;
            p0 = (k0 + l16      <= rowg) ? p0 : 0.0f;
            p1 = (k0 + 16 + l16 <= rowg) ? p1 : 0.0f;
        }
        lsum[r] += p0 + p1;
        const int row = r + half * 8;
        pb[row][l16]      = (bf16_t)p0;
        pb[row][16 + l16] = (bf16_t)p1;
    }

    // A_p: 16x32 (keys along K). same-wave LDS RAW -> hw-ordered via DScnt
    v16bf ap;
    {
        const v8bf plo = *(const v8bf*)&pb[l16][half * 16];
        const v8bf phi = *(const v8bf*)&pb[l16][half * 16 + 8];
        #pragma unroll
        for (int h = 0; h < 8; ++h) { ap[h] = plo[h]; ap[8 + h] = phi[h]; }
    }

    // ---- P @ X : 8 chunks of 16 columns -------------------------------
    #pragma unroll
    for (int cc = 0; cc < 8; ++cc) {
        const v16bf bx = *(const v16bf*)(xbase + (long)cc * 16 * SEQ + k0);
        acc[cc] = __builtin_amdgcn_wmma_f32_16x16x32_bf16(
            false, ap, false, bx, (short)0, acc[cc], false, false);
    }
}

// ---------------------------------------------------------------------------
// Kernel 3: causal attention, fixed-reference-point softmax (no running max:
// scores here are O(1..5) so raw exp2 stays comfortably in fp32 range; the
// normalizing sum is accumulated per-lane and reduced once after the k loop).
// Block = (16 query rows, 1 batch), 8 waves; wave owns 128 output columns.
// ---------------------------------------------------------------------------
__global__ __launch_bounds__(256) void flash_attn(
    const bf16_t* __restrict__ Qb, const bf16_t* __restrict__ Kb,
    const bf16_t* __restrict__ XT, float* __restrict__ out)
{
    __shared__ bf16_t pbuf[8][16][32];     // per-wave P tile [row][key]
    const int lane = threadIdx.x & 31;
    const int wave = threadIdx.x >> 5;
    const int l16  = lane & 15;
    const int half = lane >> 4;
    const int qt   = blockIdx.x;           // 0..127
    const int b    = blockIdx.y;           // 0..7
    const int cs   = wave * 128;           // column slice base

    // A_q: loop-invariant 16x32 A operand (e padded 16->32 with zeros)
    v16bf aq = zero_v16bf();
    {
        if (half == 0)
            aq = *(const v16bf*)(Qb + ((long)b * SEQ + qt * 16 + l16) * EDIM);
    }

    float lsum[8];
    v8f acc[8];
    #pragma unroll
    for (int r = 0; r < 8; ++r) lsum[r] = 0.0f;
    #pragma unroll
    for (int cc = 0; cc < 8; ++cc)
        #pragma unroll
        for (int r = 0; r < 8; ++r) acc[cc][r] = 0.0f;

    const int nkb = (qt * 16 + 16 + 31) >> 5;   // 32-key blocks (>= 1)

    const bf16_t* kbase = Kb + (long)b * SEQ * EDIM + (long)l16 * EDIM;
    const bf16_t* xbase = XT + ((long)b * DMODEL + cs + l16) * SEQ + half * 16;

    // main loop: fully unmasked blocks (everything strictly below diagonal)
    for (int kb = 0; kb < nkb - 1; ++kb)
        kblock<false>(kb * 32, qt, l16, half, pbuf[wave], kbase, xbase,
                      aq, lsum, acc);
    // single partially-masked diagonal block
    kblock<true>((nkb - 1) * 32, qt, l16, half, pbuf[wave], kbase, xbase,
                 aq, lsum, acc);

    // ---- one-time row-sum reduction across the 16 lanes of each half --
    float inv[8];
    #pragma unroll
    for (int r = 0; r < 8; ++r) {
        float t = lsum[r];
        #pragma unroll
        for (int d = 1; d < 16; d <<= 1) t += __shfl_xor(t, d, 32);
        inv[r] = 1.0f / t;
    }

    // ---- epilogue: normalize, write fp32 ------------------------------
    #pragma unroll
    for (int r = 0; r < 8; ++r) {
        const int rowg = qt * 16 + r + half * 8;
        #pragma unroll
        for (int cc = 0; cc < 8; ++cc)
            out[((long)b * SEQ + rowg) * DMODEL + cs + cc * 16 + l16] =
                acc[cc][r] * inv[r];
    }
}

// ---------------------------------------------------------------------------
extern "C" void kernel_launch(void* const* d_in, const int* in_sizes, int n_in,
                              void* d_out, int out_size, void* d_ws, size_t ws_size,
                              hipStream_t stream) {
    const float* x  = (const float*)d_in[0];
    const float* wq = (const float*)d_in[1];
    const float* bq = (const float*)d_in[2];
    const float* wk = (const float*)d_in[3];
    const float* bk = (const float*)d_in[4];
    float* out = (float*)d_out;

    // workspace layout (needs ~33.6 MB):
    //   Qb : B*S*16 bf16 = 512 KB      @ 0
    //   Kb : B*S*16 bf16 = 512 KB      @ 512K
    //   XT : B*D*S bf16  = 32 MB       @ 1M
    //   Wqb: 16*1024 bf16 = 32 KB      @ 1M + 32M
    //   Wkb: 16*1024 bf16 = 32 KB      @ 1M + 32M + 32K
    char* ws = (char*)d_ws;
    bf16_t* Qb  = (bf16_t*)ws;
    bf16_t* Kb  = (bf16_t*)(ws + 524288);
    bf16_t* XT  = (bf16_t*)(ws + 1048576);
    bf16_t* Wqb = (bf16_t*)(ws + 1048576 + 33554432);
    bf16_t* Wkb = (bf16_t*)(ws + 1048576 + 33554432 + 32768);

    cvt_w<<<EDIM * DMODEL / 256, 256, 0, stream>>>(wq, wk, Wqb, Wkb);
    proj_qk<<<128, 256, 0, stream>>>(x, Wqb, bq, Wkb, bk, Qb, Kb);
    transpose_x<<<dim3(SEQ / 64, DMODEL / 64, BATCH), 256, 0, stream>>>(x, XT);
    flash_attn<<<dim3(SEQ / 16, BATCH), 256, 0, stream>>>(Qb, Kb, XT, out);
}
